// Attention_38147899523668
// MI455X (gfx1250) — matrix operand order repaired and checked by reading, not compiled, not run on hardware
//
#include <hip/hip_runtime.h>
#include <hip/hip_bf16.h>

typedef __bf16 bf16;
typedef __attribute__((ext_vector_type(8)))  __bf16 v8bf;
typedef __attribute__((ext_vector_type(16))) __bf16 v16bf;
typedef __attribute__((ext_vector_type(8)))  float  v8f;

#define B_   2
#define S_   2048
#define H_   2048
#define NH_  32
#define NKV_ 8
#define HD_  64
#define G_   4
#define M_   (B_*S_)

static __device__ __forceinline__ v8f wmma_bf16(v16bf a, v16bf b, v8f c) {
  // D = A(16x32 bf16) * B(32x16 bf16) + C(16x16 f32)
  return __builtin_amdgcn_wmma_f32_16x16x32_bf16(false, a, false, b, (short)0, c,
                                                 false, false);
}

// A fragment: 16x32 tile from row-major [16, ld] bf16 at p = element (0,0).
// Lane layout: row = lane&15; elem t: K = 16*half + t (contiguous).
static __device__ __forceinline__ v16bf ld_afrag(const bf16* p, int ld) {
  int lane = threadIdx.x & 31;
  int row = lane & 15, hf = lane >> 4;
  const bf16* q = p + (size_t)row * ld + 16 * hf;
  v8bf lo = *(const v8bf*)(q);
  v8bf hi = *(const v8bf*)(q + 8);
  v16bf a;
#pragma unroll
  for (int i = 0; i < 8; ++i) { a[i] = lo[i]; a[i + 8] = hi[i]; }
  return a;
}

// B fragment: (K=32)x(N=16). Source stored "transposed" row-major [N, ld]
// (row n holds contiguous K). p points at (n0, k0). Lane layout: col = lane&15,
// elem t: K = 16*half + t (contiguous).
static __device__ __forceinline__ v16bf ld_bfrag(const bf16* p, size_t ld) {
  int lane = threadIdx.x & 31;
  int col = lane & 15, hf = lane >> 4;
  return *(const v16bf*)(p + (size_t)col * ld + 16 * hf);
}

// LDS byte-offset of a pointer into a __shared__ array (addrspace(3) cast).
static __device__ __forceinline__ unsigned lds_off(const bf16* p) {
  return (unsigned)(size_t)(__attribute__((address_space(3))) const bf16*)p;
}

__global__ void k_cvt(const float* __restrict__ s, bf16* __restrict__ d, int n) {
  int i = blockIdx.x * blockDim.x + threadIdx.x;
  if (i < n) d[i] = (bf16)s[i];
}

// Fused QKV projection + bias + RoPE. One wave computes 32 rows x 64 cols
// (one full head) = 2x4 C fragments: every RoPE pair (d, d+32) is wave-local,
// each B fragment is reused across both row groups, each A fragment across
// 4 column groups (8 WMMAs per 12 b128 loads).
// cg: [0,32)=Q head cg, [32,40)=K head cg-32, [40,48)=V head cg-40.
__global__ void __launch_bounds__(256)
k_proj(const bf16* __restrict__ xb,
       const bf16* __restrict__ Wqb, const bf16* __restrict__ Wkb,
       const bf16* __restrict__ Wvb,
       const float* __restrict__ bq, const float* __restrict__ bk,
       const float* __restrict__ bv,
       const float* __restrict__ fcos, const float* __restrict__ fsin,
       bf16* __restrict__ Qr, bf16* __restrict__ Kr, bf16* __restrict__ Vt)
{
  int wave = (blockIdx.x * blockDim.x + threadIdx.x) >> 5;
  int lane = threadIdx.x & 31;
  int col = lane & 15, hf = lane >> 4;
  int mtile = wave / 48, cg = wave % 48;
  int m0 = mtile * 32;

  const bf16* W; const float* bias; int mat, head;
  if (cg < 32)      { mat = 0; W = Wqb; bias = bq; head = cg; }
  else if (cg < 40) { mat = 1; W = Wkb; bias = bk; head = cg - 32; }
  else              { mat = 2; W = Wvb; bias = bv; head = cg - 40; }

  const bf16* arow = xb + (size_t)m0 * H_;
  const bf16* bp0 = W + (size_t)(head * 64 +  0 + col) * H_ + 16 * hf;
  const bf16* bp1 = W + (size_t)(head * 64 + 16 + col) * H_ + 16 * hf;
  const bf16* bp2 = W + (size_t)(head * 64 + 32 + col) * H_ + 16 * hf;
  const bf16* bp3 = W + (size_t)(head * 64 + 48 + col) * H_ + 16 * hf;

  v8f c00 = {}, c01 = {}, c02 = {}, c03 = {};
  v8f c10 = {}, c11 = {}, c12 = {}, c13 = {};
#pragma unroll 2
  for (int k0 = 0; k0 < H_; k0 += 32) {
    __builtin_prefetch(arow + (size_t)(lane & 15) * H_ + k0 + 256, 0, 1);
    __builtin_prefetch(bp0 + k0 + 256, 0, 1);
    v16bf a0 = ld_afrag(arow + k0, H_);
    v16bf a1 = ld_afrag(arow + 16 * (size_t)H_ + k0, H_);
    v16bf b0 = *(const v16bf*)(bp0 + k0);
    v16bf b1 = *(const v16bf*)(bp1 + k0);
    v16bf b2 = *(const v16bf*)(bp2 + k0);
    v16bf b3 = *(const v16bf*)(bp3 + k0);
    c00 = wmma_bf16(a0, b0, c00);
    c01 = wmma_bf16(a0, b1, c01);
    c02 = wmma_bf16(a0, b2, c02);
    c03 = wmma_bf16(a0, b3, c03);
    c10 = wmma_bf16(a1, b0, c10);
    c11 = wmma_bf16(a1, b1, c11);
    c12 = wmma_bf16(a1, b2, c12);
    c13 = wmma_bf16(a1, b3, c13);
  }
  float bb0 = bias[head * 64 +  0 + col];
  float bb1 = bias[head * 64 + 16 + col];
  float bb2 = bias[head * 64 + 32 + col];
  float bb3 = bias[head * 64 + 48 + col];

  int b  = m0 / S_;
  int s0 = m0 - b * S_;

  if (mat < 2) {
    // RoPE pairs: (frag0, frag2) at d=col, (frag1, frag3) at d=16+col
    bf16* dstBase = (mat == 0)
        ? (Qr + (size_t)(b * NH_  + head) * S_ * HD_)
        : (Kr + (size_t)(b * NKV_ + head) * S_ * HD_);
    int dA = col, dB = 16 + col;
#pragma unroll
    for (int g = 0; g < 2; ++g) {
      v8f f0 = g ? c10 : c00, f1 = g ? c11 : c01;
      v8f f2 = g ? c12 : c02, f3 = g ? c13 : c03;
#pragma unroll
      for (int r = 0; r < 8; ++r) {
        int s = s0 + 16 * g + r + 8 * hf;
        float csA = fcos[s * 32 + dA], snA = fsin[s * 32 + dA];
        float csB = fcos[s * 32 + dB], snB = fsin[s * 32 + dB];
        float yA  = f0[r] + bb0, yA2 = f2[r] + bb2;
        float yB  = f1[r] + bb1, yB2 = f3[r] + bb3;
        bf16* row = dstBase + (size_t)s * HD_;
        row[dA]      = (bf16)(yA  * csA - yA2 * snA);
        row[dA + 32] = (bf16)(yA2 * csA + yA  * snA);
        row[dB]      = (bf16)(yB  * csB - yB2 * snB);
        row[dB + 32] = (bf16)(yB2 * csB + yB  * snB);
      }
    }
  } else {
    // V: store transposed [B, NKV, HD, S] for contiguous P*V B-fragments
    size_t base = (size_t)(b * NKV_ + head) * HD_;
#pragma unroll
    for (int g = 0; g < 2; ++g) {
      v8f f0 = g ? c10 : c00, f1 = g ? c11 : c01;
      v8f f2 = g ? c12 : c02, f3 = g ? c13 : c03;
#pragma unroll
      for (int r = 0; r < 8; ++r) {
        int s = s0 + 16 * g + r + 8 * hf;
        Vt[(base +  0 + col) * S_ + s] = (bf16)(f0[r] + bb0);
        Vt[(base + 16 + col) * S_ + s] = (bf16)(f1[r] + bb1);
        Vt[(base + 32 + col) * S_ + s] = (bf16)(f2[r] + bb2);
        Vt[(base + 48 + col) * S_ + s] = (bf16)(f3[r] + bb3);
      }
    }
  }
}

// Flash attention: one wave per (b, head, 16-row q-tile). K tiles are staged
// to LDS via async global->LDS DMA (ASYNCcnt path), P is re-swizzled
// C-layout -> A-layout via LDS.
__global__ void __launch_bounds__(256)
k_attn(const bf16* __restrict__ Qr, const bf16* __restrict__ Kr,
       const bf16* __restrict__ Vt, bf16* __restrict__ attn)
{
  __shared__ __align__(128) bf16 smp[8][16 * 32];   // P staging, 1KB/wave
  __shared__ __align__(128) bf16 smk[8][32 * 64];   // K tile,    4KB/wave
  int widx = threadIdx.x >> 5;
  int wave = (blockIdx.x * blockDim.x + threadIdx.x) >> 5;
  int lane = threadIdx.x & 31;
  int col = lane & 15, hf = lane >> 4;

  int b    = wave / (NH_ * (S_ / 16));
  int rem  = wave % (NH_ * (S_ / 16));
  int head = rem / (S_ / 16);
  int q0   = (rem % (S_ / 16)) * 16;
  int kvh  = head / G_;

  const bf16* qb = Qr + ((size_t)(b * NH_ + head) * S_ + q0) * HD_;
  v16bf qa0 = ld_afrag(qb, HD_);        // d 0..31
  v16bf qa1 = ld_afrag(qb + 32, HD_);   // d 32..63

  const bf16* kb_p = Kr + (size_t)(b * NKV_ + kvh) * S_ * HD_;
  const bf16* vb_p = Vt + (size_t)(b * NKV_ + kvh) * HD_ * S_;

  unsigned smk_base = lds_off(&smk[widx][0]);

  float mrow[8], lrow[8];
#pragma unroll
  for (int r = 0; r < 8; ++r) { mrow[r] = -1e30f; lrow[r] = 0.f; }
  v8f o0 = {}, o1 = {}, o2 = {}, o3 = {};

  const int kend = q0 + 16;             // causal upper bound (exclusive)
  for (int kb = 0; kb < kend; kb += 32) {
    const bf16* kp = kb_p + (size_t)kb * HD_;   // 32 keys x 64 d, row-major

    // Previous iteration's LDS reads must finish before we overwrite smk/smp
    // (DS and ASYNC counters are mutually unordered).
    asm volatile("s_wait_dscnt 0x0" ::: "memory");
    // Async copy 4KB K tile into LDS: 8 x (32 lanes x 16B).
#pragma unroll
    for (int i = 0; i < 8; ++i) {
      unsigned      lofs = smk_base + (unsigned)((i * 32 + lane) * 16);
      unsigned long gadr = (unsigned long)(size_t)kp + (size_t)(i * 32 + lane) * 16;
      asm volatile("global_load_async_to_lds_b128 %0, %1, off"
                   :: "v"(lofs), "v"(gadr) : "memory");
    }
    asm volatile("s_wait_asynccnt 0x0" ::: "memory");

    // K B-fragments out of LDS: col = key(local), elems = d contiguous.
    const bf16* kl = &smk[widx][0];
    v16bf kf00 = *(const v16bf*)(kl + (     col) * 64 +  0 + 16 * hf);
    v16bf kf01 = *(const v16bf*)(kl + (     col) * 64 + 32 + 16 * hf);
    v16bf kf10 = *(const v16bf*)(kl + (16 + col) * 64 +  0 + 16 * hf);
    v16bf kf11 = *(const v16bf*)(kl + (16 + col) * 64 + 32 + 16 * hf);
    v8f s0 = {}, s1 = {};
    s0 = wmma_bf16(qa0, kf00, s0);
    s0 = wmma_bf16(qa1, kf01, s0);
    s1 = wmma_bf16(qa0, kf10, s1);
    s1 = wmma_bf16(qa1, kf11, s1);

#pragma unroll
    for (int r = 0; r < 8; ++r) {
      int q = q0 + r + 8 * hf;
      float x0 = s0[r] * 0.125f; if (kb + col > q)      x0 = -1e30f;
      float x1 = s1[r] * 0.125f; if (kb + 16 + col > q) x1 = -1e30f;
      float mx = fmaxf(x0, x1);
#pragma unroll
      for (int msk = 1; msk < 16; msk <<= 1) mx = fmaxf(mx, __shfl_xor(mx, msk, 32));
      float mn    = fmaxf(mrow[r], mx);
      float alpha = __expf(mrow[r] - mn);
      mrow[r] = mn;
      float p0 = __expf(x0 - mn);
      float p1 = __expf(x1 - mn);
      float ps = p0 + p1;
#pragma unroll
      for (int msk = 1; msk < 16; msk <<= 1) ps += __shfl_xor(ps, msk, 32);
      lrow[r] = lrow[r] * alpha + ps;
      o0[r] *= alpha; o1[r] *= alpha; o2[r] *= alpha; o3[r] *= alpha;
      smp[widx][(r + 8 * hf) * 32 + col]      = (bf16)p0;
      smp[widx][(r + 8 * hf) * 32 + 16 + col] = (bf16)p1;
    }
    asm volatile("s_wait_dscnt 0x0" ::: "memory");
    // Re-read P in A-fragment layout (per-wave LDS, DS pipe keeps order)
    v8bf plo = *(const v8bf*)&smp[widx][col * 32 + 16 * hf];
    v8bf phi = *(const v8bf*)&smp[widx][col * 32 + 16 * hf + 8];
    v16bf pa;
#pragma unroll
    for (int i = 0; i < 8; ++i) { pa[i] = plo[i]; pa[i + 8] = phi[i]; }

    v16bf vf0 = ld_bfrag(vb_p + kb,                   (size_t)S_);
    v16bf vf1 = ld_bfrag(vb_p + 16 * (size_t)S_ + kb, (size_t)S_);
    v16bf vf2 = ld_bfrag(vb_p + 32 * (size_t)S_ + kb, (size_t)S_);
    v16bf vf3 = ld_bfrag(vb_p + 48 * (size_t)S_ + kb, (size_t)S_);
    o0 = wmma_bf16(pa, vf0, o0);
    o1 = wmma_bf16(pa, vf1, o1);
    o2 = wmma_bf16(pa, vf2, o2);
    o3 = wmma_bf16(pa, vf3, o3);
  }

  bf16* orow = attn + (size_t)(b * S_ + q0) * (NH_ * HD_) + head * HD_;
#pragma unroll
  for (int r = 0; r < 8; ++r) {
    float inv = 1.0f / lrow[r];
    bf16* p = orow + (size_t)(r + 8 * hf) * (NH_ * HD_);
    p[col]      = (bf16)(o0[r] * inv);
    p[16 + col] = (bf16)(o1[r] * inv);
    p[32 + col] = (bf16)(o2[r] * inv);
    p[48 + col] = (bf16)(o3[r] * inv);
  }
}

// out = attn(bf16) @ Wo^T(bf16), f32 output. 32x64 tile per wave:
// 8 WMMAs per (2 A-frag + 4 B-frag) loads.
__global__ void __launch_bounds__(256)
k_oproj(const bf16* __restrict__ attn, const bf16* __restrict__ Wob,
        float* __restrict__ out)
{
  int wave = (blockIdx.x * blockDim.x + threadIdx.x) >> 5;
  int lane = threadIdx.x & 31;
  int col = lane & 15, hf = lane >> 4;
  int mtile = wave >> 5;        // 32 n-tiles of 64
  int nt    = wave & 31;
  int m0 = mtile * 32, n0 = nt * 64;
  const bf16* arow = attn + (size_t)m0 * H_;
  const bf16* bp0 = Wob + (size_t)(n0 +  0 + col) * H_ + 16 * hf;
  const bf16* bp1 = Wob + (size_t)(n0 + 16 + col) * H_ + 16 * hf;
  const bf16* bp2 = Wob + (size_t)(n0 + 32 + col) * H_ + 16 * hf;
  const bf16* bp3 = Wob + (size_t)(n0 + 48 + col) * H_ + 16 * hf;
  v8f c00 = {}, c01 = {}, c02 = {}, c03 = {};
  v8f c10 = {}, c11 = {}, c12 = {}, c13 = {};
#pragma unroll 2
  for (int k0 = 0; k0 < H_; k0 += 32) {
    __builtin_prefetch(arow + (size_t)(lane & 15) * H_ + k0 + 256, 0, 1);
    __builtin_prefetch(bp0 + k0 + 256, 0, 1);
    v16bf a0 = ld_afrag(arow + k0, H_);
    v16bf a1 = ld_afrag(arow + 16 * (size_t)H_ + k0, H_);
    v16bf b0 = *(const v16bf*)(bp0 + k0);
    v16bf b1 = *(const v16bf*)(bp1 + k0);
    v16bf b2 = *(const v16bf*)(bp2 + k0);
    v16bf b3 = *(const v16bf*)(bp3 + k0);
    c00 = wmma_bf16(a0, b0, c00);
    c01 = wmma_bf16(a0, b1, c01);
    c02 = wmma_bf16(a0, b2, c02);
    c03 = wmma_bf16(a0, b3, c03);
    c10 = wmma_bf16(a1, b0, c10);
    c11 = wmma_bf16(a1, b1, c11);
    c12 = wmma_bf16(a1, b2, c12);
    c13 = wmma_bf16(a1, b3, c13);
  }
#pragma unroll
  for (int g = 0; g < 2; ++g) {
    v8f f0 = g ? c10 : c00, f1 = g ? c11 : c01;
    v8f f2 = g ? c12 : c02, f3 = g ? c13 : c03;
#pragma unroll
    for (int r = 0; r < 8; ++r) {
      float* orow = out + (size_t)(m0 + 16 * g + r + 8 * hf) * H_ + n0 + col;
      orow[0]  = f0[r];
      orow[16] = f1[r];
      orow[32] = f2[r];
      orow[48] = f3[r];
    }
  }
}

extern "C" void kernel_launch(void* const* d_in, const int* in_sizes, int n_in,
                              void* d_out, int out_size, void* d_ws, size_t ws_size,
                              hipStream_t stream)
{
  const float* x  = (const float*)d_in[0];
  const float* fc = (const float*)d_in[1];
  const float* fs = (const float*)d_in[2];
  // d_in[3] = mask (unused: causal mask applied analytically)
  const float* Wq = (const float*)d_in[4];
  const float* bq = (const float*)d_in[5];
  const float* Wk = (const float*)d_in[6];
  const float* bk = (const float*)d_in[7];
  const float* Wv = (const float*)d_in[8];
  const float* bv = (const float*)d_in[9];
  const float* Wo = (const float*)d_in[10];
  float* out = (float*)d_out;

  char* ws = (char*)d_ws;
  size_t off = 0;
  auto alloc = [&](size_t bytes) {
    char* p = ws + off;
    off += (bytes + 255) & ~(size_t)255;
    return p;
  };
  bf16* xb   = (bf16*)alloc((size_t)M_ * H_ * 2);
  bf16* Wqb  = (bf16*)alloc((size_t)NH_  * HD_ * H_ * 2);
  bf16* Wkb  = (bf16*)alloc((size_t)NKV_ * HD_ * H_ * 2);
  bf16* Wvb  = (bf16*)alloc((size_t)NKV_ * HD_ * H_ * 2);
  bf16* Wob  = (bf16*)alloc((size_t)H_ * NH_ * HD_ * 2);
  bf16* Qr   = (bf16*)alloc((size_t)B_ * NH_  * S_ * HD_ * 2);
  bf16* Kr   = (bf16*)alloc((size_t)B_ * NKV_ * S_ * HD_ * 2);
  bf16* Vt   = (bf16*)alloc((size_t)B_ * NKV_ * HD_ * S_ * 2);
  bf16* attn = (bf16*)alloc((size_t)M_ * NH_ * HD_ * 2);
  (void)ws_size; (void)in_sizes; (void)n_in; (void)out_size;

  auto cvt = [&](const float* s, bf16* d, int n) {
    k_cvt<<<(n + 255) / 256, 256, 0, stream>>>(s, d, n);
  };
  cvt(x,  xb,  M_ * H_);
  cvt(Wq, Wqb, NH_  * HD_ * H_);
  cvt(Wk, Wkb, NKV_ * HD_ * H_);
  cvt(Wv, Wvb, NKV_ * HD_ * H_);
  cvt(Wo, Wob, H_ * NH_ * HD_);

  // 128 m-tiles(32 rows) x 48 head-groups = 6144 waves -> 768 blocks of 8 waves
  k_proj<<<768, 256, 0, stream>>>(xb, Wqb, Wkb, Wvb, bq, bk, bv, fc, fs,
                                  Qr, Kr, Vt);
  // B*NH*(S/16) = 8192 waves -> 1024 blocks
  k_attn<<<1024, 256, 0, stream>>>(Qr, Kr, Vt, attn);
  // 128 m-tiles(32 rows) x 32 n-tiles(64 wide) = 4096 waves -> 512 blocks
  k_oproj<<<512, 256, 0, stream>>>(attn, Wob, out);
}
